// SocNet_14998025798270
// MI455X (gfx1250) — matrix operand order repaired and checked by reading, not compiled, not run on hardware
//
#include <hip/hip_runtime.h>
#include <hip/hip_bf16.h>

#define NN 100000
#define NE 1600000
#define EPSF 1e-5f

typedef __attribute__((ext_vector_type(16))) _Float16 v16h;
typedef __attribute__((ext_vector_type(8)))  float    v8f;

// ---------------- WMMA helpers (gfx1250 layouts) ----------------

__device__ __forceinline__ v8f wmma32(v16h a, v16h b, v8f c) {
  return __builtin_amdgcn_wmma_f32_16x16x32_f16(false, a, false, b, (short)0, c, false, false);
}

// A-matrix 16x32 f16 from an LDS f32 tile [16][ldk], k-window at kbase.
// lane<16: M=lane, K=0..15; lane>=16: M=lane-16, K=16..31
__device__ __forceinline__ v16h build_a_lds(const float* tile, int ldk, int kbase, int lane) {
  int m = lane & 15;
  int khalf = (lane & 16) ? 16 : 0;
  v16h a;
#pragma unroll
  for (int i = 0; i < 16; ++i) {
    int k = i + khalf;
    a[i] = (_Float16)tile[m * ldk + kbase + k];
  }
  return a;
}

// B-matrix 32x16 f16 from global f32 W[k*ldn + n] (row-major [K][Nout]).
// lane<16: N=lane, K=0..15 ; lane>=16: N=lane-16, K=16..31 (within the 32-k chunk)
__device__ __forceinline__ v16h build_b_glb(const float* W, int ldn, int ntile,
                                            int kbase, int kvalid, int nvalid, int lane) {
  int n = (lane & 15) + ntile * 16;
  int koff = (lane & 16) ? 16 : 0;
  v16h b;
#pragma unroll
  for (int i = 0; i < 16; ++i) {
    int k = kbase + koff + i;
    float v = (k < kvalid && n < nvalid) ? W[k * ldn + n] : 0.0f;
    b[i] = (_Float16)v;
  }
  return b;
}

// D f32 16x16: vgpr r -> M = r + (lane>=16 ? 8 : 0), N = (lane&15) + 16*ntile
__device__ __forceinline__ void store_tile_relu(float* out, int ldn, int node, int ntile,
                                                v8f d, int lane) {
  int mh = (lane & 16) ? 8 : 0;
  int n = (lane & 15) + ntile * 16;
#pragma unroll
  for (int r = 0; r < 8; ++r) {
    float v = d[r]; v = v > 0.f ? v : 0.f;
    out[(size_t)(node + mh + r) * ldn + n] = v;
  }
}

// ---------------- InstanceNorm ----------------

__global__ void in_stats_k(const float* __restrict__ x, float* stats, int ncols, int total) {
  __shared__ float ssum[256];
  __shared__ float ssq[256];
  int tid = threadIdx.x;
  int gid = blockIdx.x * blockDim.x + tid;
  int stride = gridDim.x * blockDim.x;   // multiple of 256 -> channel fixed per thread
  float s = 0.f, q = 0.f;
  for (int i = gid; i < total; i += stride) {
    float v = x[i];
    s += v; q += v * v;
  }
  ssum[tid] = s; ssq[tid] = q;
  __syncthreads();
  for (int st = 128; st >= ncols; st >>= 1) {
    if (tid < st) { ssum[tid] += ssum[tid + st]; ssq[tid] += ssq[tid + st]; }
    __syncthreads();
  }
  if (tid < ncols) {
    atomicAdd(&stats[tid], ssum[tid]);
    atomicAdd(&stats[32 + tid], ssq[tid]);
  }
}

__global__ void in_finalize_k(float* stats, int ncols, float invn) {
  int c = threadIdx.x;
  if (c < ncols) {
    float mean = stats[c] * invn;
    float var  = stats[32 + c] * invn - mean * mean;
    stats[64 + c] = mean;
    stats[96 + c] = rsqrtf(var + EPSF);
  }
}

__global__ void in_apply_k(const float* __restrict__ x, const float* __restrict__ stats,
                           float* __restrict__ xn, int cmask, int total) {
  int i = blockIdx.x * blockDim.x + threadIdx.x;
  if (i < total) {
    int c = i & cmask;
    xn[i] = (x[i] - stats[64 + c]) * stats[96 + c];
  }
}

// ---------------- gcn_norm ----------------

__global__ void deg_k(const int* __restrict__ col, const float* __restrict__ ea,
                      float* __restrict__ deg, int E) {
  int e = blockIdx.x * blockDim.x + threadIdx.x;
  if (e < E) atomicAdd(&deg[col[e]], ea[e]);
}

__global__ void dinv_k(float* deg, int n) {  // in place: deg -> d^{-1/2}
  int i = blockIdx.x * blockDim.x + threadIdx.x;
  if (i < n) {
    float d = deg[i];
    deg[i] = d > 0.f ? rsqrtf(d) : 0.f;
  }
}

__global__ void normw_k(const int* __restrict__ row, const int* __restrict__ col,
                        const float* __restrict__ ea, const float* __restrict__ dinv,
                        float* __restrict__ nw, int E) {
  int e = blockIdx.x * blockDim.x + threadIdx.x;
  if (e < E) nw[e] = dinv[row[e]] * ea[e] * dinv[col[e]];
}

// ---------------- TAGConv hop scatters ----------------

__global__ void hop32_k(const float* __restrict__ hp, const float* __restrict__ nw,
                        const int* __restrict__ row, const int* __restrict__ col,
                        float* __restrict__ hn, int E) {
  int gid = blockIdx.x * blockDim.x + threadIdx.x;
  int e = gid >> 3;
  if (e < E) {
    int sub = gid & 7;
    float w = nw[e];
    int r = row[e], c = col[e];
    float4 v = ((const float4*)hp)[(size_t)r * 8 + sub];
    float* dst = hn + (size_t)c * 32 + sub * 4;
    atomicAdd(dst + 0, w * v.x);
    atomicAdd(dst + 1, w * v.y);
    atomicAdd(dst + 2, w * v.z);
    atomicAdd(dst + 3, w * v.w);
  }
}

__global__ void hop1_k(const float* __restrict__ hp, const float* __restrict__ nw,
                       const int* __restrict__ row, const int* __restrict__ col,
                       float* __restrict__ hn, int E) {
  int e = blockIdx.x * blockDim.x + threadIdx.x;
  if (e < E) atomicAdd(&hn[col[e]], nw[e] * hp[row[e]]);
}

// ---------------- TAGConv output GEMM: relu(sum_k h_k @ W_k + b) ----------------
// one wave per block, 4 node-tiles (64 nodes) per wave; weights hoisted

__global__ void tag_gemm_k(const float* __restrict__ h0, const float* __restrict__ h1,
                           const float* __restrict__ h2, const float* __restrict__ h3,
                           const float* __restrict__ w0, const float* __restrict__ w1,
                           const float* __restrict__ w2, const float* __restrict__ w3,
                           const float* __restrict__ bias, float* __restrict__ out, int idim) {
  __shared__ alignas(16) float tile[512];
  int lane = threadIdx.x;
  const float* hs[4] = {h0, h1, h2, h3};
  const float* ws[4] = {w0, w1, w2, w3};

  // hoist weights into registers
  v16h B[4][2];
  if (idim == 32) {
#pragma unroll
    for (int kc = 0; kc < 4; ++kc) {
      B[kc][0] = build_b_glb(ws[kc], 32, 0, 0, 32, 32, lane);
      B[kc][1] = build_b_glb(ws[kc], 32, 1, 0, 32, 32, lane);
    }
  } else {  // idim==1: single chunk; B row k = ws[k] (1x32)
    int n = lane & 15;
    int koff = (lane & 16) ? 16 : 0;
#pragma unroll
    for (int i = 0; i < 16; ++i) {
      float v0 = 0.f, v1 = 0.f;
      if (koff == 0 && i < 4) { v0 = ws[i][n]; v1 = ws[i][n + 16]; }
      B[0][0][i] = (_Float16)v0;
      B[0][1][i] = (_Float16)v1;
    }
    // zero the padded tile once (cols 4..31 stay zero across tiles)
    for (int i = lane; i < 512; i += 32) tile[i] = 0.f;
  }
  float bn0 = bias[lane & 15];
  float bn1 = bias[(lane & 15) + 16];

  for (int t = 0; t < 4; ++t) {
    int node = blockIdx.x * 64 + t * 16;
    if (node >= NN) break;   // uniform branch: EXEC stays full at WMMA sites
    v8f c0, c1;
#pragma unroll
    for (int r = 0; r < 8; ++r) { c0[r] = bn0; c1[r] = bn1; }
    if (idim == 32) {
#pragma unroll
      for (int kc = 0; kc < 4; ++kc) {
        __syncthreads();
        const float4* src = (const float4*)(hs[kc] + (size_t)node * 32);
        float4* dst = (float4*)tile;
#pragma unroll
        for (int i = 0; i < 4; ++i) dst[lane + i * 32] = src[lane + i * 32];
        __syncthreads();
        v16h a = build_a_lds(tile, 32, 0, lane);
        c0 = wmma32(a, B[kc][0], c0);
        c1 = wmma32(a, B[kc][1], c1);
      }
    } else {
      __syncthreads();
      if (lane < 16) {
#pragma unroll
        for (int k = 0; k < 4; ++k) tile[lane * 32 + k] = hs[k][node + lane];
      }
      __syncthreads();
      v16h a = build_a_lds(tile, 32, 0, lane);
      c0 = wmma32(a, B[0][0], c0);
      c1 = wmma32(a, B[0][1], c1);
    }
    store_tile_relu(out, 32, node, 0, c0, lane);
    store_tile_relu(out, 32, node, 1, c1, lane);
  }
}

// ---------------- FC layer: relu(x @ W + b), Nout = 32 or 1 ----------------
// one wave per block, 4 node-tiles per wave; weights hoisted

__global__ void fc_gemm_k(const float* __restrict__ x, const float* __restrict__ W,
                          const float* __restrict__ bias, float* __restrict__ out, int nout) {
  __shared__ alignas(16) float tile[512];
  int lane = threadIdx.x;
  v16h bb0, bb1;
  float bn0, bn1 = 0.f;
  if (nout == 32) {
    bb0 = build_b_glb(W, 32, 0, 0, 32, 32, lane);
    bb1 = build_b_glb(W, 32, 1, 0, 32, 32, lane);
    bn0 = bias[lane & 15];
    bn1 = bias[(lane & 15) + 16];
  } else {
    bb0 = build_b_glb(W, 1, 0, 0, 32, 1, lane);
    bn0 = ((lane & 15) == 0) ? bias[0] : 0.f;
  }

  for (int t = 0; t < 4; ++t) {
    int node = blockIdx.x * 64 + t * 16;
    if (node >= NN) break;
    __syncthreads();
    const float4* src = (const float4*)(x + (size_t)node * 32);
    float4* dst = (float4*)tile;
#pragma unroll
    for (int i = 0; i < 4; ++i) dst[lane + i * 32] = src[lane + i * 32];
    __syncthreads();
    v16h a = build_a_lds(tile, 32, 0, lane);
    if (nout == 32) {
      v8f c0, c1;
#pragma unroll
      for (int r = 0; r < 8; ++r) { c0[r] = bn0; c1[r] = bn1; }
      c0 = wmma32(a, bb0, c0);
      c1 = wmma32(a, bb1, c1);
      store_tile_relu(out, 32, node, 0, c0, lane);
      store_tile_relu(out, 32, node, 1, c1, lane);
    } else {
      v8f c0;
#pragma unroll
      for (int r = 0; r < 8; ++r) c0[r] = bn0;
      c0 = wmma32(a, bb0, c0);
      if ((lane & 15) == 0) {
        int mh = (lane & 16) ? 8 : 0;
#pragma unroll
        for (int r = 0; r < 8; ++r) {
          float v = c0[r];
          out[node + mh + r] = v > 0.f ? v : 0.f;
        }
      }
    }
  }
}

// ---------------- fused edge MLP ----------------
// relu + LayerNorm on a 16x32 D tile; LN params live in LDS

__device__ __forceinline__ void relu_ln(v8f c0, v8f c1, const float* g,
                                        const float* b, float* act, int lane) {
  __syncthreads();
  {
    int mh = (lane & 16) ? 8 : 0;
    int n = lane & 15;
#pragma unroll
    for (int r = 0; r < 8; ++r) {
      float v0 = c0[r]; v0 = v0 > 0.f ? v0 : 0.f;
      float v1 = c1[r]; v1 = v1 > 0.f ? v1 : 0.f;
      act[(mh + r) * 32 + n]      = v0;
      act[(mh + r) * 32 + n + 16] = v1;
    }
  }
  __syncthreads();
  int rowp = lane >> 1;      // 2 lanes per row, 16 features each
  int half = lane & 1;
  float vals[16];
  float s = 0.f, q = 0.f;
#pragma unroll
  for (int j = 0; j < 16; ++j) {
    float v = act[rowp * 32 + half * 16 + j];
    vals[j] = v; s += v; q += v * v;
  }
  s += __shfl_xor(s, 1, 32);
  q += __shfl_xor(q, 1, 32);
  float mean = s * (1.0f / 32.0f);
  float var  = q * (1.0f / 32.0f) - mean * mean;
  float inv  = rsqrtf(var + EPSF);
  __syncthreads();
#pragma unroll
  for (int j = 0; j < 16; ++j) {
    int f = half * 16 + j;
    act[rowp * 32 + f] = (vals[j] - mean) * inv * g[f] + b[f];
  }
  __syncthreads();
}

// one wave per block; 8 edge-tiles (128 edges) per wave; all weights hoisted
__global__ void edge_mlp_k(const float* __restrict__ x, const int* __restrict__ row,
                           const int* __restrict__ col, const float* __restrict__ ea_in,
                           const float* __restrict__ w1, const float* __restrict__ b1,
                           const float* __restrict__ g1, const float* __restrict__ bb1,
                           const float* __restrict__ w2, const float* __restrict__ b2,
                           const float* __restrict__ g2, const float* __restrict__ bb2,
                           const float* __restrict__ w3, const float* __restrict__ b3,
                           float* __restrict__ ea_out, int nd) {
  __shared__ alignas(16) float sin_t[16 * 96];
  __shared__ alignas(16) float act[16 * 32];
  __shared__ float lnp[128];
  int lane = threadIdx.x;
  int kin = (nd == 32) ? 96 : 32;   // K padded to multiple of 32
  int kv = 2 * nd + 1;              // valid input width (65 or 3)
  int nchunks = kin >> 5;

  // hoist weights / params (once per wave)
  v16h B1[3][2], B2[2];
#pragma unroll
  for (int kc = 0; kc < 3; ++kc) {
    B1[kc][0] = build_b_glb(w1, 32, 0, kc * 32, kv, 32, lane);
    B1[kc][1] = build_b_glb(w1, 32, 1, kc * 32, kv, 32, lane);
  }
  B2[0] = build_b_glb(w2, 32, 0, 0, 32, 32, lane);
  B2[1] = build_b_glb(w2, 32, 1, 0, 32, 32, lane);
  float b1n0 = b1[lane & 15], b1n1 = b1[(lane & 15) + 16];
  float b2n0 = b2[lane & 15], b2n1 = b2[(lane & 15) + 16];
  float b3v = b3[0];
  float w3r[16];
#pragma unroll
  for (int j = 0; j < 16; ++j) w3r[j] = w3[(lane & 1) * 16 + j];
  lnp[lane] = g1[lane];
  lnp[32 + lane] = bb1[lane];
  lnp[64 + lane] = g2[lane];
  lnp[96 + lane] = bb2[lane];
  // zero-pad the input staging tile once (padded cols never rewritten)
  for (int i = lane; i < 16 * 96; i += 32) sin_t[i] = 0.f;
  __syncthreads();

  for (int t = 0; t < 8; ++t) {
    int e0 = blockIdx.x * 128 + t * 16;
    if (t < 7) {
      __builtin_prefetch(row + e0 + 16);
      __builtin_prefetch(col + e0 + 16);
    }
    __syncthreads();
    // gather [x[row] | x[col] | ea] -> sin_t
    {
      int m = lane & 15;
      int e = e0 + m;
      if (nd == 32) {
        if (lane < 16) {
          int r = row[e];
          const float4* src = (const float4*)(x + (size_t)r * 32);
          float4* dst = (float4*)(sin_t + m * 96);
#pragma unroll
          for (int i = 0; i < 8; ++i) dst[i] = src[i];
          sin_t[m * 96 + 64] = ea_in[e];
        } else {
          int c = col[e];
          const float4* src = (const float4*)(x + (size_t)c * 32);
          float4* dst = (float4*)(sin_t + m * 96 + 32);
#pragma unroll
          for (int i = 0; i < 8; ++i) dst[i] = src[i];
        }
      } else {
        if (lane < 16) {
          int r = row[e];
          sin_t[m * 32 + 0] = x[r];
          sin_t[m * 32 + 2] = ea_in[e];
        } else {
          int c = col[e];
          sin_t[m * 32 + 1] = x[c];
        }
      }
    }
    __syncthreads();

    // L1: 16 x kin @ kv x 32
    v8f c0, c1;
#pragma unroll
    for (int r = 0; r < 8; ++r) { c0[r] = b1n0; c1[r] = b1n1; }
#pragma unroll
    for (int kc = 0; kc < 3; ++kc) {
      if (kc < nchunks) {
        v16h a = build_a_lds(sin_t, kin, kc * 32, lane);
        c0 = wmma32(a, B1[kc][0], c0);
        c1 = wmma32(a, B1[kc][1], c1);
      }
    }
    relu_ln(c0, c1, lnp, lnp + 32, act, lane);

    // L2: 16x32 @ 32x32
#pragma unroll
    for (int r = 0; r < 8; ++r) { c0[r] = b2n0; c1[r] = b2n1; }
    {
      v16h a = build_a_lds(act, 32, 0, lane);
      c0 = wmma32(a, B2[0], c0);
      c1 = wmma32(a, B2[1], c1);
    }
    relu_ln(c0, c1, lnp + 64, lnp + 96, act, lane);

    // L3: 32 -> 1 dot per edge (2 lanes per row), + relu
    int rowp = lane >> 1;
    int half = lane & 1;
    float s = 0.f;
#pragma unroll
    for (int j = 0; j < 16; ++j) s += act[rowp * 32 + half * 16 + j] * w3r[j];
    s += __shfl_xor(s, 1, 32);
    s += b3v;
    s = s > 0.f ? s : 0.f;
    if (half == 0) ea_out[e0 + rowp] = s;
  }
}

// ---------------- host driver ----------------

extern "C" void kernel_launch(void* const* d_in, const int* in_sizes, int n_in,
                              void* d_out, int out_size, void* d_ws, size_t ws_size,
                              hipStream_t stream) {
  const int N = NN, E = NE;

  int ix = -1, iei = -1, iea = -1;
  int pidx[128]; int np = 0;
  for (int i = 0; i < n_in; ++i) {
    if (in_sizes[i] == N && ix < 0) ix = i;
    else if (in_sizes[i] == 2 * E && iei < 0) iei = i;
    else if (in_sizes[i] == E && iea < 0) iea = i;
    else if (np < 128) pidx[np++] = i;
  }
  if (ix < 0 || iei < 0 || iea < 0 || np < 92) return;

  const float* x_in = (const float*)d_in[ix];
  const int* ei = (const int*)d_in[iei];
  const int* rowp = ei;
  const int* colp = ei + E;
  const float* ea0 = (const float*)d_in[iea];

  float* ws = (float*)d_ws;
  size_t o = 0;
  float* xn  = ws + o; o += (size_t)N * 32;
  float* h1  = ws + o; o += (size_t)N * 32;
  float* h2  = ws + o; o += (size_t)N * 32;
  float* h3  = ws + o; o += (size_t)N * 32;
  float* xA  = ws + o; o += (size_t)N * 32;
  float* xB  = ws + o; o += (size_t)N * 32;
  float* deg = ws + o; o += (size_t)N;
  float* nw  = ws + o; o += (size_t)E;
  float* eaA = ws + o; o += (size_t)E;
  float* eaB = ws + o; o += (size_t)E;
  float* stats = ws + o; o += 128;
  if (ws_size < o * sizeof(float)) return;

  float* xfinal = (float*)d_out;       // N
  float* eafinal = (float*)d_out + N;  // E

  const float* x_cur = x_in;
  const float* ea_cur = ea0;

  for (int p = 0; p < 4; ++p) {
    int base = p * 23;
    const float* ec_l1_b = (const float*)d_in[pidx[base + 0]];
    const float* ec_l1_w = (const float*)d_in[pidx[base + 1]];
    const float* ec_l2_b = (const float*)d_in[pidx[base + 2]];
    const float* ec_l2_w = (const float*)d_in[pidx[base + 3]];
    const float* ec_l3_b = (const float*)d_in[pidx[base + 4]];
    const float* ec_l3_w = (const float*)d_in[pidx[base + 5]];
    const float* ln1_b = (const float*)d_in[pidx[base + 6]];
    const float* ln1_g = (const float*)d_in[pidx[base + 7]];
    const float* ln2_b = (const float*)d_in[pidx[base + 8]];
    const float* ln2_g = (const float*)d_in[pidx[base + 9]];
    const float* fcb[4], * fcw[4];
    for (int j = 0; j < 4; ++j) {
      fcb[j] = (const float*)d_in[pidx[base + 10 + 2 * j]];
      fcw[j] = (const float*)d_in[pidx[base + 11 + 2 * j]];
    }
    const float* tag_b = (const float*)d_in[pidx[base + 18]];
    const float* tagw[4];
    for (int k = 0; k < 4; ++k) tagw[k] = (const float*)d_in[pidx[base + 19 + k]];

    int idim  = (p == 0) ? 1 : 32;  // TAGConv input width
    int ndout = (p < 3) ? 32 : 1;   // fc output width / edge-MLP node dim

    // InstanceNorm
    hipMemsetAsync(stats, 0, 128 * sizeof(float), stream);
    in_stats_k<<<256, 256, 0, stream>>>(x_cur, stats, idim, N * idim);
    in_finalize_k<<<1, 32, 0, stream>>>(stats, idim, 1.0f / (float)N);
    in_apply_k<<<(N * idim + 255) / 256, 256, 0, stream>>>(x_cur, stats, xn, idim - 1, N * idim);

    // gcn_norm on current edge_attr
    hipMemsetAsync(deg, 0, (size_t)N * sizeof(float), stream);
    deg_k<<<(E + 255) / 256, 256, 0, stream>>>(colp, ea_cur, deg, E);
    dinv_k<<<(N + 255) / 256, 256, 0, stream>>>(deg, N);
    normw_k<<<(E + 255) / 256, 256, 0, stream>>>(rowp, colp, ea_cur, deg, nw, E);

    // K=3 hop scatters
    const float* hprev = xn;
    float* harr[3] = {h1, h2, h3};
    for (int k = 0; k < 3; ++k) {
      float* hn = harr[k];
      hipMemsetAsync(hn, 0, (size_t)N * idim * sizeof(float), stream);
      if (idim == 32)
        hop32_k<<<(E * 8 + 255) / 256, 256, 0, stream>>>(hprev, nw, rowp, colp, hn, E);
      else
        hop1_k<<<(E + 255) / 256, 256, 0, stream>>>(hprev, nw, rowp, colp, hn, E);
      hprev = hn;
    }

    // TAGConv GEMM + relu -> xB   (4 tiles / wave)
    tag_gemm_k<<<(N + 63) / 64, 32, 0, stream>>>(xn, h1, h2, h3,
                                                 tagw[0], tagw[1], tagw[2], tagw[3],
                                                 tag_b, xB, idim);

    // FC chain (h1 reused as temp; hop buffers dead now)
    fc_gemm_k<<<(N + 63) / 64, 32, 0, stream>>>(xB, fcw[0], fcb[0], h1, 32);
    fc_gemm_k<<<(N + 63) / 64, 32, 0, stream>>>(h1, fcw[1], fcb[1], xB, 32);
    fc_gemm_k<<<(N + 63) / 64, 32, 0, stream>>>(xB, fcw[2], fcb[2], h1, 32);
    float* x_next = (p < 3) ? xA : xfinal;
    fc_gemm_k<<<(N + 63) / 64, 32, 0, stream>>>(h1, fcw[3], fcb[3], x_next, ndout);

    // fused edge MLP -> next edge_attr   (8 tiles / wave)
    float* ea_next = (p == 0) ? eaA : (p == 1) ? eaB : (p == 2) ? eaA : eafinal;
    edge_mlp_k<<<E / 128, 32, 0, stream>>>(x_next, rowp, colp, ea_cur,
                                           ec_l1_w, ec_l1_b, ln1_g, ln1_b,
                                           ec_l2_w, ec_l2_b, ln2_g, ln2_b,
                                           ec_l3_w, ec_l3_b, ea_next, ndout);

    x_cur = x_next;
    ea_cur = ea_next;
  }
  (void)out_size;
}
